// UniADMemory_32014686224742
// MI455X (gfx1250) — matrix operand order repaired and checked by reading, not compiled, not run on hardware
//
#include <hip/hip_runtime.h>
#include <hip/hip_bf16.h>
#include <stdint.h>

// Problem dims (fixed by the reference)
#define NTOK 196
#define BATCH 64
#define DDIM 256
#define MEMN 2048
#define NB (NTOK * BATCH) // 12544

typedef __attribute__((ext_vector_type(16))) __bf16 v16bf; // 8 VGPRs: WMMA bf16 A/B operand
typedef __attribute__((ext_vector_type(8))) float v8f;     // 8 VGPRs: WMMA f32 C/D operand
typedef __attribute__((ext_vector_type(4))) int v4i;       // 16B vector for async LDS copy

union BfVec {
    uint32_t u[8]; // 8 dwords = 16 packed bf16
    v16bf v;
};

#if defined(__has_builtin)
#if __has_builtin(__builtin_amdgcn_global_load_async_to_lds_b128) && \
    __has_builtin(__builtin_amdgcn_s_wait_asynccnt)
#define HAVE_ASYNC_LDS 1
#endif
#endif

typedef __attribute__((address_space(1))) v4i* gv4i_ptr; // global
typedef __attribute__((address_space(3))) v4i* lv4i_ptr; // LDS

__device__ __forceinline__ uint16_t f2bf(float f) {
    uint32_t u = __float_as_uint(f);
    u += 0x7FFFu + ((u >> 16) & 1u); // round-to-nearest-even
    return (uint16_t)(u >> 16);
}
__device__ __forceinline__ uint32_t packbf2(float lo, float hi) {
    return (uint32_t)f2bf(lo) | ((uint32_t)f2bf(hi) << 16);
}

// ---------------------------------------------------------------------------
// Precompute 1: memory-side projections (all raw, no centering needed:
// cov is recovered algebraically as qf.kf - 256*qmean*kmean).
//   Kb  [M][D] bf16 : ch_memory @ ch_wk^T
//   KFb [M][D] bf16 : sp_mem_flat @ sp_wk^T
//   Vt  [D][M] bf16 : (ch_memory @ ch_wv^T)^T   (transposed -> WMMA B pairs contiguous)
//   VFt [D][M] bf16 : (sp_mem_flat @ sp_wv^T)^T
//   k_mean/k_var [M] f32 (ddof=1)
// ---------------------------------------------------------------------------
__global__ __launch_bounds__(256) void proj_kernel(
    const float* __restrict__ ch_mem, const float* __restrict__ ch_wk,
    const float* __restrict__ ch_wv, const float* __restrict__ sp_mem,
    const float* __restrict__ sp_wk, const float* __restrict__ sp_wv,
    uint16_t* __restrict__ Kb, uint16_t* __restrict__ KFb,
    uint16_t* __restrict__ Vt, uint16_t* __restrict__ VFt,
    float* __restrict__ k_mean, float* __restrict__ k_var) {
    __shared__ float mc[DDIM], ms[DDIM];
    __shared__ float rA[DDIM], rB[DDIM];
    const int row = blockIdx.x, c = threadIdx.x;
    mc[c] = ch_mem[row * DDIM + c];
    ms[c] = sp_mem[row * DDIM + c];
    __syncthreads();
    float kc = 0.f, vc = 0.f, kf = 0.f, vf = 0.f;
#pragma unroll 4
    for (int d = 0; d < DDIM; ++d) {
        float a = mc[d], b = ms[d];
        kc = fmaf(a, ch_wk[c * DDIM + d], kc);
        vc = fmaf(a, ch_wv[c * DDIM + d], vc);
        kf = fmaf(b, sp_wk[c * DDIM + d], kf);
        vf = fmaf(b, sp_wv[c * DDIM + d], vf);
    }
    Kb[row * DDIM + c] = f2bf(kc);
    KFb[row * DDIM + c] = f2bf(kf);
    Vt[c * MEMN + row] = f2bf(vc);
    VFt[c * MEMN + row] = f2bf(vf);
    rA[c] = kf;
    rB[c] = kf * kf;
    __syncthreads();
    for (int s = 128; s > 0; s >>= 1) {
        if (c < s) {
            rA[c] += rA[c + s];
            rB[c] += rB[c + s];
        }
        __syncthreads();
    }
    if (c == 0) {
        float mean = rA[0] * (1.0f / DDIM);
        float var = (rB[0] - (float)DDIM * mean * mean) * (1.0f / (DDIM - 1));
        k_mean[row] = mean;
        k_var[row] = var;
    }
}

// ---------------------------------------------------------------------------
// Precompute 2: query-side projections (raw Qb, QFb, q_mean/q_var).
// ---------------------------------------------------------------------------
__global__ __launch_bounds__(256) void q_kernel(
    const float* __restrict__ x, const float* __restrict__ ch_wq,
    const float* __restrict__ sp_wq, uint16_t* __restrict__ Qb,
    uint16_t* __restrict__ QFb, float* __restrict__ q_mean,
    float* __restrict__ q_var) {
    __shared__ float xr[DDIM];
    __shared__ float rA[DDIM], rB[DDIM];
    const int row = blockIdx.x, c = threadIdx.x;
    xr[c] = x[row * DDIM + c];
    __syncthreads();
    float q = 0.f, qf = 0.f;
#pragma unroll 4
    for (int d = 0; d < DDIM; ++d) {
        float a = xr[d];
        q = fmaf(a, ch_wq[c * DDIM + d], q);
        qf = fmaf(a, sp_wq[c * DDIM + d], qf);
    }
    Qb[row * DDIM + c] = f2bf(q);
    QFb[row * DDIM + c] = f2bf(qf);
    rA[c] = qf;
    rB[c] = qf * qf;
    __syncthreads();
    for (int s = 128; s > 0; s >>= 1) {
        if (c < s) {
            rA[c] += rA[c + s];
            rB[c] += rB[c + s];
        }
        __syncthreads();
    }
    if (c == 0) {
        float mean = rA[0] * (1.0f / DDIM);
        float var = (rB[0] - (float)DDIM * mean * mean) * (1.0f / (DDIM - 1));
        q_mean[row] = mean;
        q_var[row] = var;
    }
}

// ---------------------------------------------------------------------------
// Main fused kernel, single pass. Scores here are tiny (|s_c| ~ 0.3,
// |ssim| <= ~1) so exp() without max-shift is numerically safe in f32 and
// softmax is shift-invariant -> no max pass, no accumulator rescaling.
// Block = 128 threads = 4 wave32s, one 16-query tile, m-tile of 64 memories.
// Wave w: score columns [membase+16w, +16), output columns [64w, 64w+64).
// ---------------------------------------------------------------------------
__global__ __launch_bounds__(128) void attn_kernel(
    const uint32_t* __restrict__ Qb32, const uint32_t* __restrict__ QFb32,
    const uint32_t* __restrict__ Kb32, const uint32_t* __restrict__ KFb32,
    const uint32_t* __restrict__ Vt32, const uint32_t* __restrict__ VFt32,
    const float* __restrict__ k_mean, const float* __restrict__ k_var,
    const float* __restrict__ q_mean, const float* __restrict__ q_var,
    float* __restrict__ out) {
    __shared__ float sc_buf[16][64];    // channel score tile (raw dot, f32)
    __shared__ float ss_buf[16][64];    // spatial raw-dot tile (f32)
    __shared__ uint32_t wc_buf[16][32]; // bf16x2 channel weights
    __shared__ uint32_t ws_buf[16][32]; // bf16x2 spatial weights
    __shared__ float red[16][8][2];
    __shared__ float lrow_c[16], lrow_s[16];
    __shared__ float qm_sh[16], qv_sh[16];
    __shared__ float km_sh[MEMN], kv_sh[MEMN];

    const int tid = threadIdx.x;
    const int lane = tid & 31;
    const int wid = tid >> 5; // 0..3
    const int nidx = lane & 15;
    const int hh = lane >> 4; // half-wave select (ISA K-split)
    const int qbase = blockIdx.x * 16;
    const int r8 = tid >> 3;      // row handled in the softmax stage
    const int c8 = (tid & 7) * 8; // first of 8 columns handled

#if HAVE_ASYNC_LDS
    // gfx1250 async global->LDS copy for the k statistics (16 KB)
    for (int i = tid * 4; i < MEMN; i += 128 * 4) {
        __builtin_amdgcn_global_load_async_to_lds_b128(
            (gv4i_ptr)(k_mean + i), (lv4i_ptr)(&km_sh[i]), 0, 0);
        __builtin_amdgcn_global_load_async_to_lds_b128(
            (gv4i_ptr)(k_var + i), (lv4i_ptr)(&kv_sh[i]), 0, 0);
    }
    __builtin_amdgcn_s_wait_asynccnt(0);
#else
    for (int i = tid; i < MEMN; i += 128) {
        km_sh[i] = k_mean[i];
        kv_sh[i] = k_var[i];
    }
#endif
    if (tid < 16) {
        qm_sh[tid] = q_mean[qbase + tid];
        qv_sh[tid] = q_var[qbase + tid];
        lrow_c[tid] = 0.f;
        lrow_s[tid] = 0.f;
    }

    // Q (channel) / QF (spatial) held in registers as WMMA A-operands.
    // A 16x32 bf16 layout: lane(m=lane&15, hh): V0..3 hold K=8hh+{0..7},
    // V4..7 hold K=16+8hh+{0..7}; bf16 pairs are contiguous dwords.
    BfVec aq[8], af[8];
    {
        const int qrow = qbase + nidx;
#pragma unroll
        for (int ck = 0; ck < 8; ++ck) {
#pragma unroll
            for (int j = 0; j < 8; ++j) {
                int k = (j < 4) ? (8 * hh + 2 * j) : (16 + 8 * hh + 2 * (j - 4));
                int e = qrow * DDIM + ck * 32 + k; // even
                aq[ck].u[j] = Qb32[e >> 1];
                af[ck].u[j] = QFb32[e >> 1];
            }
        }
    }
    __syncthreads();

    const float inv16 = 0.0625f;       // 1/sqrt(D)
    const float inv255 = 1.0f / 255.f; // 1/(HW-1)

    v8f oc[4], os[4];
#pragma unroll
    for (int t = 0; t < 4; ++t) {
        oc[t] = (v8f){0.f, 0.f, 0.f, 0.f, 0.f, 0.f, 0.f, 0.f};
        os[t] = (v8f){0.f, 0.f, 0.f, 0.f, 0.f, 0.f, 0.f, 0.f};
    }
    float sumc = 0.f, sums = 0.f; // per-thread softmax partial sums (regs)

    for (int mt = 0; mt < MEMN / 64; ++mt) {
        const int membase = mt * 64;
        // ---- score GEMMs: S = Q(16x256) . K_slice^T, double-buffered B ----
        const int kdw = ((membase + wid * 16 + nidx) * DDIM >> 1) + 4 * hh;
        BfVec bb[2], bf2[2];
#pragma unroll
        for (int j = 0; j < 8; ++j) { // prefetch chunk 0
            bb[0].u[j] = Kb32[kdw + (j < 4 ? j : j + 4)];
            bf2[0].u[j] = KFb32[kdw + (j < 4 ? j : j + 4)];
        }
        v8f accA = {0.f, 0.f, 0.f, 0.f, 0.f, 0.f, 0.f, 0.f};
        v8f accB = {0.f, 0.f, 0.f, 0.f, 0.f, 0.f, 0.f, 0.f};
#pragma unroll
        for (int ck = 0; ck < 8; ++ck) {
            const int cur = ck & 1, nxt = cur ^ 1;
            if (ck < 7) { // prefetch chunk ck+1 while WMMA consumes ck
                const int base = kdw + (ck + 1) * 16;
#pragma unroll
                for (int j = 0; j < 8; ++j) {
                    bb[nxt].u[j] = Kb32[base + (j < 4 ? j : j + 4)];
                    bf2[nxt].u[j] = KFb32[base + (j < 4 ? j : j + 4)];
                }
            }
            accA = __builtin_amdgcn_wmma_f32_16x16x32_bf16(
                false, aq[ck].v, false, bb[cur].v, (short)0, accA, false, false);
            accB = __builtin_amdgcn_wmma_f32_16x16x32_bf16(
                false, af[ck].v, false, bf2[cur].v, (short)0, accB, false,
                false);
        }
#pragma unroll
        for (int e = 0; e < 8; ++e) { // C/D: Ve = row e (lanes 0-15) / e+8
            sc_buf[e + 8 * hh][wid * 16 + nidx] = accA[e];
            ss_buf[e + 8 * hh][wid * 16 + nidx] = accB[e];
        }
        __syncthreads();

        // ---- softmax weights (unnormalized exp) for 8 columns of row r8 ----
        {
            const float qm = qm_sh[r8], qv = qv_sh[r8];
            float wc[8], ws_[8];
#pragma unroll
            for (int i = 0; i < 8; ++i) {
                int c = c8 + i, mem = membase + c;
                float km = km_sh[mem], kv = kv_sh[mem];
                float sc = sc_buf[r8][c] * inv16;
                // cov = ((qf.kf) - 256*qm*km)/255   (centering folded out)
                float cov = (ss_buf[r8][c] - (float)DDIM * qm * km) * inv255;
                float ssim =
                    ((2.f * qm * km + 0.01f) * (2.f * cov + 0.03f)) /
                    ((qm * qm + km * km + 0.01f) * (qv + kv + 0.03f) + 1e-8f);
                wc[i] = __expf(sc);
                ws_[i] = __expf(ssim);
                sumc += wc[i];
                sums += ws_[i];
            }
#pragma unroll
            for (int i = 0; i < 4; ++i) {
                wc_buf[r8][(c8 >> 1) + i] = packbf2(wc[2 * i], wc[2 * i + 1]);
                ws_buf[r8][(c8 >> 1) + i] = packbf2(ws_[2 * i], ws_[2 * i + 1]);
            }
        }
        __syncthreads();

        // ---- out(16 x 64-cols-of-this-wave) += W(16x64) @ V(64x64) ----
#pragma unroll
        for (int half = 0; half < 2; ++half) { // 32 memory rows per WMMA K
            BfVec wa, wsa; // A operand: weights, K = memory index
#pragma unroll
            for (int j = 0; j < 8; ++j) {
                int k = half * 32 +
                        ((j < 4) ? (8 * hh + 2 * j) : (16 + 8 * hh + 2 * (j - 4)));
                wa.u[j] = wc_buf[nidx][k >> 1];
                wsa.u[j] = ws_buf[nidx][k >> 1];
            }
            BfVec vb[2], vfb[2]; // B operand from transposed V, double-buffered
            {
                const int col0 = wid * 64 + nidx;
                const int vdw0 = ((col0 * MEMN + membase) >> 1) + half * 16 + 4 * hh;
#pragma unroll
                for (int j = 0; j < 8; ++j) {
                    vb[0].u[j] = Vt32[vdw0 + (j < 4 ? j : j + 4)];
                    vfb[0].u[j] = VFt32[vdw0 + (j < 4 ? j : j + 4)];
                }
            }
#pragma unroll
            for (int t = 0; t < 4; ++t) {
                const int cur = t & 1, nxt = cur ^ 1;
                if (t < 3) { // prefetch next column tile
                    const int col = wid * 64 + (t + 1) * 16 + nidx;
                    const int vdw =
                        ((col * MEMN + membase) >> 1) + half * 16 + 4 * hh;
#pragma unroll
                    for (int j = 0; j < 8; ++j) {
                        vb[nxt].u[j] = Vt32[vdw + (j < 4 ? j : j + 4)];
                        vfb[nxt].u[j] = VFt32[vdw + (j < 4 ? j : j + 4)];
                    }
                }
                oc[t] = __builtin_amdgcn_wmma_f32_16x16x32_bf16(
                    false, wa.v, false, vb[cur].v, (short)0, oc[t], false,
                    false);
                os[t] = __builtin_amdgcn_wmma_f32_16x16x32_bf16(
                    false, wsa.v, false, vfb[cur].v, (short)0, os[t], false,
                    false);
            }
        }
        __syncthreads();
    }

    // ---- final softmax-denominator reduction (once, not per tile) ----
    red[r8][tid & 7][0] = sumc;
    red[r8][tid & 7][1] = sums;
    __syncthreads();
    if ((tid & 7) == 0) {
        float a = 0.f, b = 0.f;
#pragma unroll
        for (int i = 0; i < 8; ++i) {
            a += red[r8][i][0];
            b += red[r8][i][1];
        }
        lrow_c[r8] = a;
        lrow_s[r8] = b;
    }
    __syncthreads();

    // ---- finalize: out = out_c/l_c + out_s/l_s ----
#pragma unroll
    for (int t = 0; t < 4; ++t) {
#pragma unroll
        for (int e = 0; e < 8; ++e) {
            int r = e + 8 * hh;
            int row = qbase + r;
            int col = wid * 64 + t * 16 + nidx;
            out[row * DDIM + col] = oc[t][e] / lrow_c[r] + os[t][e] / lrow_s[r];
        }
    }
}

// ---------------------------------------------------------------------------
extern "C" void kernel_launch(void* const* d_in, const int* in_sizes, int n_in,
                              void* d_out, int out_size, void* d_ws,
                              size_t ws_size, hipStream_t stream) {
    (void)in_sizes;
    (void)n_in;
    (void)out_size;
    (void)ws_size;
    const float* x = (const float*)d_in[0];
    const float* ch_mem = (const float*)d_in[1];
    const float* ch_wq = (const float*)d_in[2];
    const float* ch_wk = (const float*)d_in[3];
    const float* ch_wv = (const float*)d_in[4];
    const float* sp_mem = (const float*)d_in[5];
    const float* sp_wq = (const float*)d_in[6];
    const float* sp_wk = (const float*)d_in[7];
    const float* sp_wv = (const float*)d_in[8];
    float* out = (float*)d_out;

    char* ws = (char*)d_ws;
    size_t off = 0;
    auto carve = [&](size_t bytes) -> char* {
        off = (off + 255) & ~(size_t)255;
        char* p = ws + off;
        off += bytes;
        return p;
    };
    uint16_t* Kb = (uint16_t*)carve((size_t)MEMN * DDIM * 2);
    uint16_t* KFb = (uint16_t*)carve((size_t)MEMN * DDIM * 2);
    uint16_t* Vt = (uint16_t*)carve((size_t)MEMN * DDIM * 2);
    uint16_t* VFt = (uint16_t*)carve((size_t)MEMN * DDIM * 2);
    float* k_mean = (float*)carve((size_t)MEMN * 4);
    float* k_var = (float*)carve((size_t)MEMN * 4);
    uint16_t* Qb = (uint16_t*)carve((size_t)NB * DDIM * 2);
    uint16_t* QFb = (uint16_t*)carve((size_t)NB * DDIM * 2);
    float* q_mean = (float*)carve((size_t)NB * 4);
    float* q_var = (float*)carve((size_t)NB * 4);

    hipLaunchKernelGGL(proj_kernel, dim3(MEMN), dim3(256), 0, stream, ch_mem,
                       ch_wk, ch_wv, sp_mem, sp_wk, sp_wv, Kb, KFb, Vt, VFt,
                       k_mean, k_var);
    hipLaunchKernelGGL(q_kernel, dim3(NB), dim3(256), 0, stream, x, ch_wq,
                       sp_wq, Qb, QFb, q_mean, q_var);
    hipLaunchKernelGGL(attn_kernel, dim3(NB / 16), dim3(128), 0, stream,
                       (const uint32_t*)Qb, (const uint32_t*)QFb,
                       (const uint32_t*)Kb, (const uint32_t*)KFb,
                       (const uint32_t*)Vt, (const uint32_t*)VFt, k_mean, k_var,
                       q_mean, q_var, out);
}
